// LinearMoELayer_1924145348662
// MI455X (gfx1250) — compile-verified, matrix operand order checked by reading, not run on hardware
//
#include <hip/hip_runtime.h>

#define T_TOK 8192
#define DIN   2048
#define DOUT  2048
#define NEXP  8
#define BM 128
#define BN 128
#define KT 32
#define MAX_TILES 136        // 16384/128 + 8 (worst-case sum of ceils)
#define GATHER_CAP 17408     // 16384 + 8*128 padding slots

typedef __attribute__((ext_vector_type(16))) __bf16 v16bf;
typedef __attribute__((ext_vector_type(8)))  float  v8f;
typedef __attribute__((ext_vector_type(4)))  float  v4f;
typedef __attribute__((ext_vector_type(4)))  __bf16 v4bf;

// Convert 4 fp32 -> bf16 hi + bf16 lo (residual) and store to LDS as b64s.
// Lets the backend pick native v_cvt f32->bf16 (packed if available).
__device__ __forceinline__ void cvt_store4(unsigned short* __restrict__ hi,
                                           unsigned short* __restrict__ lo,
                                           int idx, float4 v) {
  v4f f = {v.x, v.y, v.z, v.w};
  v4bf h  = __builtin_convertvector(f, v4bf);          // RNE
  v4f  hf = __builtin_convertvector(h, v4f);           // exact widen
  v4bf l  = __builtin_convertvector(f - hf, v4bf);     // residual, exact-range in bf16 exp
  *(v4bf*)(hi + idx) = h;                              // idx % 4 == 0 -> 8B aligned
  *(v4bf*)(lo + idx) = l;
}

// ---------------- K0: zero ws header (importance/cnt/fillPos/numTiles/off) --
__global__ void zero_hdr_kernel(int* hdr) {
  if (threadIdx.x < 128) hdr[threadIdx.x] = 0;
}

// ---------------- K1: gate — logits, top-2, softmax, routing stats ---------
__global__ __launch_bounds__(256) void gate_kernel(
    const float* __restrict__ x, const float* __restrict__ Wg,
    float* __restrict__ scores, int* __restrict__ idxs,
    float* __restrict__ importance, int* __restrict__ cnt)
{
  __shared__ float wgl[NEXP * DIN];                     // 64 KB of 320 KB LDS
  for (int i = threadIdx.x; i < NEXP * DIN; i += 256) wgl[i] = Wg[i];
  __syncthreads();

  const int wave = threadIdx.x >> 5;
  const int lane = threadIdx.x & 31;
  const int t = blockIdx.x * 8 + wave;                  // grid = T/8 exactly

  const float4* xv = (const float4*)(x + (size_t)t * DIN);
  float acc[NEXP];
#pragma unroll
  for (int e = 0; e < NEXP; ++e) acc[e] = 0.f;

  for (int i = lane; i < DIN / 4; i += 32) {
    const float4 v = xv[i];
    const int d = i * 4;
#pragma unroll
    for (int e = 0; e < NEXP; ++e) {
      const float* w = &wgl[e * DIN + d];
      acc[e] = fmaf(v.x, w[0], acc[e]);
      acc[e] = fmaf(v.y, w[1], acc[e]);
      acc[e] = fmaf(v.z, w[2], acc[e]);
      acc[e] = fmaf(v.w, w[3], acc[e]);
    }
  }
#pragma unroll
  for (int e = 0; e < NEXP; ++e) {
#pragma unroll
    for (int off = 16; off > 0; off >>= 1)
      acc[e] += __shfl_xor(acc[e], off, 32);
  }

  if (lane == 0) {
    int i0 = 0;
#pragma unroll
    for (int e = 1; e < NEXP; ++e) if (acc[e] > acc[i0]) i0 = e;   // earliest index on ties
    int i1 = (i0 == 0) ? 1 : 0;
#pragma unroll
    for (int e = 0; e < NEXP; ++e) if (e != i0 && acc[e] > acc[i1]) i1 = e;
    const float ex = expf(acc[i1] - acc[i0]);           // <= 1, stable
    const float s0 = 1.0f / (1.0f + ex);
    const float s1 = 1.0f - s0;
    scores[2 * t] = s0;  scores[2 * t + 1] = s1;
    idxs[2 * t]   = i0;  idxs[2 * t + 1]   = i1;
    atomicAdd(&importance[i0], s0);
    atomicAdd(&importance[i1], s1);
    atomicAdd(&cnt[i0], 1);
    atomicAdd(&cnt[i1], 1);
  }
}

// ---------------- K2: offsets, tile list, pad sentinels, gate loss ---------
__global__ void setup_kernel(
    const float* __restrict__ importance, const int* __restrict__ cnt,
    int* __restrict__ off, int* __restrict__ tokList,
    int* __restrict__ tileE, int* __restrict__ tileM,
    int* __restrict__ numTiles, float* __restrict__ loss_out)
{
  if (threadIdx.x != 0 || blockIdx.x != 0) return;
  int o = 0, nt = 0;
  for (int e = 0; e < NEXP; ++e) {
    off[e] = o;
    const int c  = cnt[e];
    const int pc = (c + BM - 1) & ~(BM - 1);
    for (int m = 0; m < pc; m += BM) { tileE[nt] = e; tileM[nt] = o + m; ++nt; }
    for (int i = c; i < pc; ++i) tokList[o + i] = -1;   // pad sentinel
    o += pc;
  }
  *numTiles = nt;

  float mi = 0.f, ml = 0.f;
  for (int e = 0; e < NEXP; ++e) { mi += importance[e]; ml += (float)cnt[e]; }
  mi *= (1.0f / NEXP);  ml *= (1.0f / NEXP);
  float vi = 0.f, vl = 0.f;
  for (int e = 0; e < NEXP; ++e) {
    const float di = importance[e] - mi; vi += di * di;
    const float dl = (float)cnt[e] - ml; vl += dl * dl;
  }
  vi *= (1.0f / (NEXP - 1));  vl *= (1.0f / (NEXP - 1));   // ddof=1
  *loss_out = vi / (mi * mi + 1e-10f) + vl / (ml * ml + 1e-10f);
}

// ---------------- K3: scatter tokens into per-expert buckets ---------------
__global__ __launch_bounds__(256) void scatter_kernel(
    const float* __restrict__ scores, const int* __restrict__ idxs,
    const int* __restrict__ off, int* __restrict__ fillPos,
    int* __restrict__ tokList, float* __restrict__ scoreList)
{
  const int t = blockIdx.x * 256 + threadIdx.x;
  if (t >= T_TOK) return;
#pragma unroll
  for (int s = 0; s < 2; ++s) {
    const int e = idxs[2 * t + s];
    const int p = atomicAdd(&fillPos[e], 1);
    tokList[off[e] + p]   = t;
    scoreList[off[e] + p] = scores[2 * t + s];
  }
}

// ---------------- K4: y = s0*b[e0] + s1*b[e1]  (initializes all of y) ------
__global__ __launch_bounds__(256) void bias_init_kernel(
    const float* __restrict__ scores, const int* __restrict__ idxs,
    const float* __restrict__ b, float* __restrict__ y)
{
  const int t = blockIdx.x;
  const float s0 = scores[2 * t], s1 = scores[2 * t + 1];
  const int   i0 = idxs[2 * t],   i1 = idxs[2 * t + 1];
  const float* b0 = b + (size_t)i0 * DOUT;
  const float* b1 = b + (size_t)i1 * DOUT;
  float* yr = y + (size_t)t * DOUT;
  for (int o = threadIdx.x; o < DOUT; o += 256)
    yr[o] = fmaf(s0, b0[o], s1 * b1[o]);
}

// ---------------- K5: gathered per-expert GEMM, bf16 hi/lo x3 WMMA ---------
// 128x128 block tile, 8 waves (each 16M x 128N), KT=32, double-buffered LDS.
__global__ __launch_bounds__(256) void moe_gemm_kernel(
    const float* __restrict__ x, const float* __restrict__ W,
    const int* __restrict__ tokList, const float* __restrict__ scoreList,
    const int* __restrict__ tileE, const int* __restrict__ tileM,
    const int* __restrict__ numTiles, float* __restrict__ y)
{
  if ((int)blockIdx.x >= *numTiles) return;
  const int e     = tileE[blockIdx.x];
  const int mBase = tileM[blockIdx.x];
  const int nBase = blockIdx.y * BN;

  // 2 buffers x (128x32) x hi/lo x (A,B) = exactly 64 KB
  __shared__ __align__(64) unsigned short AhiS[2][BM * KT], AloS[2][BM * KT];
  __shared__ __align__(64) unsigned short BhiS[2][BN * KT], BloS[2][BN * KT];

  const int tid  = threadIdx.x;
  const int lane = tid & 31;
  const int half = lane >> 4;          // contiguous k order (A and B alike): lanes 16-31 hold K+16
  const int lr   = lane & 15;
  const int mSub = (tid >> 5) * 16;    // 8 waves -> 8 M sub-tiles
  const int aOff = (mSub + lr) * KT + half * 16;

  // Loop-invariant per-thread load coordinates: 4 A granules + 4 B granules
  const float* pA[4];
  const float* pB[4];
  int ldsIdx[4];
#pragma unroll
  for (int j = 0; j < 4; ++j) {
    const int g  = tid + j * 256;      // 1024 float4 granules per 128x32 tile
    const int r  = g >> 3;
    const int kc = (g & 7) << 2;
    int tok = tokList[mBase + r]; tok = (tok < 0) ? 0 : tok;  // pad rows: any valid addr
    pA[j]     = x + (size_t)tok * DIN + kc;
    pB[j]     = W + ((size_t)e * DOUT + nBase + r) * DIN + kc;
    ldsIdx[j] = r * KT + kc;
  }

  v8f cacc[8] = {};
  float4 fa[4], fb[4];

  // prologue: load + convert + store K-step 0 into buffer 0
#pragma unroll
  for (int j = 0; j < 4; ++j) { fa[j] = *(const float4*)pA[j]; fb[j] = *(const float4*)pB[j]; }
#pragma unroll
  for (int j = 0; j < 4; ++j) {
    cvt_store4(AhiS[0], AloS[0], ldsIdx[j], fa[j]);
    cvt_store4(BhiS[0], BloS[0], ldsIdx[j], fb[j]);
  }
  __syncthreads();

  const int kIters = DIN / KT;         // 64
  int buf = 0;
  for (int kt = 0; kt < kIters; ++kt) {
    // prefetch next K-step while this one computes
    if (kt + 1 < kIters) {
      const int k0 = (kt + 1) * KT;
#pragma unroll
      for (int j = 0; j < 4; ++j) {
        fa[j] = *(const float4*)(pA[j] + k0);
        fb[j] = *(const float4*)(pB[j] + k0);
      }
    }

    // compute from current buffer: fp32 emu = hi*hi + hi*lo + lo*hi
    const v16bf aH = *(const v16bf*)&AhiS[buf][aOff];
    const v16bf aL = *(const v16bf*)&AloS[buf][aOff];
#pragma unroll
    for (int nt = 0; nt < 8; ++nt) {
      const int bo = (nt * 16 + lr) * KT + half * 16;
      const v16bf bH = *(const v16bf*)&BhiS[buf][bo];
      const v16bf bL = *(const v16bf*)&BloS[buf][bo];
      cacc[nt] = __builtin_amdgcn_wmma_f32_16x16x32_bf16(false, aH, false, bH, (short)0, cacc[nt], false, false);
      cacc[nt] = __builtin_amdgcn_wmma_f32_16x16x32_bf16(false, aH, false, bL, (short)0, cacc[nt], false, false);
      cacc[nt] = __builtin_amdgcn_wmma_f32_16x16x32_bf16(false, aL, false, bH, (short)0, cacc[nt], false, false);
    }

    // convert + store prefetched data into the other buffer
    if (kt + 1 < kIters) {
      const int nb = buf ^ 1;
#pragma unroll
      for (int j = 0; j < 4; ++j) {
        cvt_store4(AhiS[nb], AloS[nb], ldsIdx[j], fa[j]);
        cvt_store4(BhiS[nb], BloS[nb], ldsIdx[j], fb[j]);
      }
      __syncthreads();
      buf ^= 1;
    }
  }

  // Epilogue. C layout: VGPR j, lanes 0-15 -> M=j, lanes 16-31 -> M=8+j, N=lane%16
#pragma unroll
  for (int j = 0; j < 8; ++j) {
    const int row = mSub + half * 8 + j;
    const int tok = tokList[mBase + row];
    if (tok >= 0) {
      const float s = scoreList[mBase + row];
      float* yr = y + (size_t)tok * DOUT + nBase;
#pragma unroll
      for (int nt = 0; nt < 8; ++nt)
        atomicAdd(&yr[nt * 16 + lr], s * cacc[nt][j]);
    }
  }
}

extern "C" void kernel_launch(void* const* d_in, const int* in_sizes, int n_in,
                              void* d_out, int out_size, void* d_ws, size_t ws_size,
                              hipStream_t stream)
{
  (void)in_sizes; (void)n_in; (void)out_size; (void)ws_size;
  const float* x  = (const float*)d_in[0];
  const float* Wg = (const float*)d_in[1];
  const float* W  = (const float*)d_in[2];
  const float* b  = (const float*)d_in[3];
  float* y = (float*)d_out;
  float* loss_out = y + (size_t)T_TOK * DOUT;   // gate_loss is last output elem

  char* ws = (char*)d_ws;
  float* importance = (float*)(ws + 0);     // 8 f32
  int*   cnt        = (int*)(ws + 64);      // 8 i32
  int*   fillPos    = (int*)(ws + 128);     // 8 i32
  int*   numTiles   = (int*)(ws + 192);     // 1 i32
  int*   off        = (int*)(ws + 256);     // 8 i32
  size_t p = 512;
  float* scores     = (float*)(ws + p); p += (size_t)2 * T_TOK * 4;
  int*   idxs       = (int*)(ws + p);   p += (size_t)2 * T_TOK * 4;
  int*   tokList    = (int*)(ws + p);   p += (size_t)GATHER_CAP * 4;
  float* scoreList  = (float*)(ws + p); p += (size_t)GATHER_CAP * 4;
  int*   tileE      = (int*)(ws + p);   p += 512 * 4;
  int*   tileM      = (int*)(ws + p);   p += 512 * 4;

  zero_hdr_kernel<<<1, 128, 0, stream>>>((int*)ws);
  gate_kernel<<<T_TOK / 8, 256, 0, stream>>>(x, Wg, scores, idxs, importance, cnt);
  setup_kernel<<<1, 1, 0, stream>>>(importance, cnt, off, tokList, tileE, tileM, numTiles, loss_out);
  scatter_kernel<<<T_TOK / 256, 256, 0, stream>>>(scores, idxs, off, fillPos, tokList, scoreList);
  bias_init_kernel<<<T_TOK, 256, 0, stream>>>(scores, idxs, b, y);
  dim3 g(MAX_TILES, DOUT / BN);
  moe_gemm_kernel<<<g, 256, 0, stream>>>(x, W, tokList, scoreList, tileE, tileM, numTiles, y);
}
